// CorrelationModule_33904471834987
// MI455X (gfx1250) — matrix operand order repaired and checked by reading, not compiled, not run on hardware
//
#include <hip/hip_runtime.h>
#include <hip/hip_bf16.h>
#include <math.h>

// ---------------------------------------------------------------------------
// CorrelationModule for MI455X (gfx1250, wave32, WMMA).
//
// Pipeline:
//   k0 : zero a 512B "zero page" (one pixel's 256 f16 channels) in workspace.
//   k1 : fused norm + transpose: stage 256ch x 32px f32 tile in LDS, reduce
//        L2 norms in-LDS, write normalized f16 features layout [b][h*w][c].
//        (Single pass over x: halves HBM traffic vs separate norm kernel.)
//   k2 : per (b, y) row, 7 waves; each wave owns an 8-pixel tile and runs
//        9 (dy) x 8 (K-chunks) chained v_wmma_f32_16x16x32_f16 ops:
//        C[16x16] = A(16 cur pixels x 256ch) * B(256ch x 16 post pixels),
//        C[m][n] = corr at dx = n-m-4. OOB / missing post-frame lanes read
//        the zero page (address predication). All B loads for a dy issue
//        ahead of the WMMA chain (sched_barrier) for latency overlap; band
//        entries scatter to LDS via index-predicated stores (dump slot).
//        56-thread argmax + narrow-Gaussian soft-argmax -> (h, v, conf).
//
// Workspace: zero page (512B) + xm f16 [64][3136][256] (~98 MB).
// ---------------------------------------------------------------------------

#define BT 64
#define CCH 256
#define HH 56
#define WW 56
#define DD 4
#define KK 9
#define HW (HH * WW)

typedef _Float16 v8h  __attribute__((ext_vector_type(8)));
typedef _Float16 v16h __attribute__((ext_vector_type(16)));
typedef float    v8f  __attribute__((ext_vector_type(8)));

// ---- k0: zero page init ----------------------------------------------------
__global__ void corr_zeropage_kernel(float* __restrict__ zp) {
    zp[threadIdx.x] = 0.0f;   // 128 threads x 4B = 512B
}

// ---- k1: fused norm + transpose + f16 convert ------------------------------
// Block: 32x8 threads handles one (b, 32-pixel) tile, all 256 channels.
__global__ void __launch_bounds__(256)
corr_prep_kernel(const float* __restrict__ x, _Float16* __restrict__ xm) {
    __shared__ float tile[CCH][33];   // [channel][pixel], padded
    __shared__ float part[8][33];
    __shared__ float invs[32];

    const int b      = blockIdx.y;
    const int hwbase = blockIdx.x * 32;       // 98 tiles (3136 = 98*32)
    const int tx = threadIdx.x;               // 0..31 (pixel within tile)
    const int ty = threadIdx.y;               // 0..7

    // Stage 256x32 f32 block; coalesced 128B rows per wave.
    const float* src = x + ((size_t)b * CCH) * HW + hwbase;
#pragma unroll
    for (int i = 0; i < 32; ++i) {
        const int c = ty + 8 * i;
        tile[c][tx] = src[(size_t)c * HW + tx];
    }
    __syncthreads();

    // Partial sum of squares: 8 threads per pixel, 32 channels each.
    float ss = 0.0f;
#pragma unroll
    for (int k = 0; k < 32; ++k) {
        const float v = tile[ty * 32 + k][tx];
        ss = fmaf(v, v, ss);
    }
    part[ty][tx] = ss;
    __syncthreads();
    if (ty == 0) {
        float t = 0.0f;
#pragma unroll
        for (int j = 0; j < 8; ++j) t += part[j][tx];
        invs[tx] = 1.0f / fmaxf(sqrtf(t), 1e-12f);
    }
    __syncthreads();

    // Scaled transposed f16 store: [b][hw][c], coalesced along c.
#pragma unroll
    for (int i = 0; i < 4; ++i) {
        const int hw_l = ty + 8 * i;
        const float sc = invs[hw_l];
        _Float16* dst = xm + ((size_t)b * HW + hwbase + hw_l) * CCH;
#pragma unroll
        for (int j = 0; j < 8; ++j) {
            const int c = tx + 32 * j;
            dst[c] = (_Float16)(tile[c][hw_l] * sc);
        }
    }
}

// ---- k2: correlation volume via WMMA + soft-argmax -------------------------
__global__ void __launch_bounds__(224)
corr_wmma_kernel(const _Float16* __restrict__ xm,
                 const _Float16* __restrict__ zeropg,
                 float* __restrict__ out) {
    const int b    = blockIdx.x;     // 0..63  (frame)
    const int y    = blockIdx.y;     // 0..55  (row)
    const int tid  = threadIdx.x;    // 0..223 (7 waves)
    const int wave = tid >> 5;       // 0..6  -> x-tile
    const int lane = tid & 31;
    const int n    = lane & 15;      // A row index / B column index
    const int hi   = lane >> 4;      // K half selector

    // Flat band volume + 1 dump slot for index-predicated stores.
    __shared__ float r[WW * KK * KK + 1];

    const int x0 = wave * 8;

    // ---- A: current frame, row y, pixels x0..x0+15 (clamped), all 256 ch.
    // Per ISA A layout: lane holds K = {8*hi..8*hi+7} and {16+8*hi..16+8*hi+7}
    // per 32-channel chunk -> two contiguous 16B loads.
    const int xa = x0 + n < WW ? x0 + n : WW - 1;
    const _Float16* pa = xm + ((size_t)b * HW + (size_t)y * WW + xa) * CCH;
    v16h a[8];
#pragma unroll
    for (int kc = 0; kc < 8; ++kc) {
        const v8h lo = *(const v8h*)(pa + kc * 32 + hi * 8);
        const v8h up = *(const v8h*)(pa + kc * 32 + 16 + hi * 8);
#pragma unroll
        for (int e = 0; e < 8; ++e) { a[kc][e] = lo[e]; a[kc][e + 8] = up[e]; }
    }

    // ---- B: post frame = b-1 within the 8-frame segment, zero otherwise.
    const bool have_post = (b & 7) != 0;
    const _Float16* post = xm + (size_t)(b - 1) * HW * CCH;
    const int xp = x0 - DD + n;                 // B column pixel
    const bool col_ok = (xp >= 0) && (xp < WW);

    for (int dyi = 0; dyi < KK; ++dyi) {
        const int yp = y + dyi - DD;
        const bool ok = have_post && (yp >= 0) && (yp < HH) && col_ok;
        // Address predication: OOB lanes read the 512B zero page.
        const _Float16* pb =
            ok ? (post + ((size_t)yp * WW + xp) * CCH) : zeropg;

        // Issue all 8 B chunks (16 x b128) ahead of the WMMA chain.
        // B layout matches A: lane holds K = {8*hi..8*hi+7} and {16+8*hi..16+8*hi+7}.
        v16h bm[8];
#pragma unroll
        for (int kc = 0; kc < 8; ++kc) {
            const v8h blo = *(const v8h*)(pb + kc * 32 + hi * 8);
            const v8h bup = *(const v8h*)(pb + kc * 32 + 16 + hi * 8);
#pragma unroll
            for (int e = 0; e < 8; ++e) { bm[kc][e] = blo[e]; bm[kc][e + 8] = bup[e]; }
        }

        // Prefetch next dy's B row while this chain runs (global_prefetch_b8).
        if (dyi + 1 < KK) {
            const int yp2 = yp + 1;
            const bool ok2 = have_post && (yp2 >= 0) && (yp2 < HH) && col_ok;
            const _Float16* pn =
                ok2 ? (post + ((size_t)yp2 * WW + xp) * CCH) : zeropg;
            __builtin_prefetch(pn, 0, 1);
        }

#if __has_builtin(__builtin_amdgcn_sched_barrier)
        __builtin_amdgcn_sched_barrier(0);  // keep loads above the WMMA chain
#endif

        v8f acc = {0.f, 0.f, 0.f, 0.f, 0.f, 0.f, 0.f, 0.f};
#pragma unroll
        for (int kc = 0; kc < 8; ++kc)
            acc = __builtin_amdgcn_wmma_f32_16x16x32_f16(
                false, a[kc], false, bm[kc], (short)0, acc, false, false);

        // C layout: lane (hi*16+n) VGPR j holds C[j + 8*hi][n].
        // Useful rows m=0..7 live in lanes 0..15; dx index = n - m.
        // Index-predicated stores: invalid entries go to the dump slot.
        if (hi == 0) {
#pragma unroll
            for (int j = 0; j < 8; ++j) {
                const int dxi = n - j;
                const bool valid = (dxi >= 0) && (dxi <= 8);
                const int widx = valid
                    ? ((x0 + j) * (KK * KK) + dyi * KK + dxi)
                    : (WW * KK * KK);
                r[widx] = acc[j];
            }
        }
    }
    __syncthreads();

    // ---- per-pixel post-processing: argmax + soft-argmax ------------------
    if (tid < WW) {
        const float* rr = &r[tid * KK * KK];
        float cmax = rr[0];
        int   idx  = 0;
        for (int kk = 1; kk < KK * KK; ++kk) {
            const float v = rr[kk];
            if (v > cmax) { cmax = v; idx = kk; }
        }
        const float hidx = (float)(idx / KK - DD);   // row offset of argmax
        const float vidx = (float)(idx % KK - DD);   // col offset of argmax

        // flat = gauss * r * BETA ; p = flat / (sum + 1e-12)
        float s = 1e-12f;
        for (int kk = 0; kk < KK * KK; ++kk) {
            const float gy = (float)(kk / KK - DD);
            const float gx = (float)(kk % KK - DD);
            const float cy = gy - hidx, cx = gx - vidx;
            s += __expf(-(cy * cy + cx * cx) * 50.0f) * rr[kk] * 10.0f;
        }
        const float inv_s = 1.0f / s;
        float hc = 0.0f, vc = 0.0f;
        for (int kk = 0; kk < KK * KK; ++kk) {
            const float gy = (float)(kk / KK - DD);
            const float gx = (float)(kk % KK - DD);
            const float cy = gy - hidx, cx = gx - vidx;
            const float p =
                __expf(-(cy * cy + cx * cx) * 50.0f) * rr[kk] * 10.0f * inv_s;
            hc = fmaf(p, gy, hc);
            vc = fmaf(p, gx, vc);
        }

        const size_t obase = (size_t)b * 3 * HW + (size_t)y * WW + tid;
        out[obase]          = hc;    // channel 0: h_cord
        out[obase + HW]     = vc;    // channel 1: v_cord
        out[obase + 2 * HW] = cmax;  // channel 2: conf
    }
}

// ---------------------------------------------------------------------------
extern "C" void kernel_launch(void* const* d_in, const int* in_sizes, int n_in,
                              void* d_out, int out_size, void* d_ws, size_t ws_size,
                              hipStream_t stream) {
    (void)in_sizes; (void)n_in; (void)out_size; (void)ws_size;

    const float* x = (const float*)d_in[0];
    float* out = (float*)d_out;

    // Workspace: [zero page: 512B][xm: 64*3136*256 f16]
    char* ws = (char*)d_ws;
    float* zp = (float*)ws;
    _Float16* xm = (_Float16*)(ws + 512);

    // k0: zero page (re-written every call; deterministic, graph-safe)
    corr_zeropage_kernel<<<1, 128, 0, stream>>>(zp);

    // k1: fused norm + transpose -> normalized f16 features, channel-last
    dim3 gP(HW / 32, BT);             // (98, 64)
    dim3 bP(32, 8);
    corr_prep_kernel<<<gP, bP, 0, stream>>>(x, xm);

    // k2: WMMA correlation + soft-argmax
    dim3 gC(BT, HH);                  // (64, 56)
    corr_wmma_kernel<<<gC, 224, 0, stream>>>(xm, (const _Float16*)zp, out);
}
